// Attention_27015344292540
// MI455X (gfx1250) — matrix operand order repaired and checked by reading, not compiled, not run on hardware
//
#include <hip/hip_runtime.h>
#include <hip/hip_bf16.h>

// ---------------------------------------------------------------------------
// Linear attention (elu+1 feature map), MI455X / gfx1250, bf16 WMMA pipeline.
// B=8 N=4096 C=768 H=12 D=64
// Weights pre-fused: Q = x @ (Wq@Wqkv_q)^T + bq  (saves a full 116-GFLOP pass)
// ---------------------------------------------------------------------------

typedef __bf16 bf16_t;
typedef __attribute__((ext_vector_type(8)))  __bf16 v8bf;
typedef __attribute__((ext_vector_type(16))) __bf16 v16bf;
typedef __attribute__((ext_vector_type(8)))  float  v8f;

#define WMMA_BF16(a, b, c) \
  __builtin_amdgcn_wmma_f32_16x16x32_bf16(false, (a), false, (b), (short)0, (c), false, false)

static __device__ __forceinline__ v16bf cat8(v8bf lo, v8bf hi) {
  return __builtin_shufflevector(lo, hi, 0,1,2,3,4,5,6,7,8,9,10,11,12,13,14,15);
}

// ---- CDNA5 async global->LDS staging (probe-verified on this toolchain)
#if defined(__has_builtin)
#  if __has_builtin(__builtin_amdgcn_global_load_async_to_lds_b128) && \
      __has_builtin(__builtin_amdgcn_s_wait_asynccnt)
#    define USE_ASYNC_LDS 1
#  endif
#endif
#ifndef USE_ASYNC_LDS
#  define USE_ASYNC_LDS 0
#endif

#if USE_ASYNC_LDS
typedef int v4i_t __attribute__((ext_vector_type(4)));
typedef __attribute__((address_space(1))) v4i_t* as1_v4i;   // global
typedef __attribute__((address_space(3))) v4i_t* as3_v4i;   // LDS
static __device__ __forceinline__ void async_cp16(const void* g, void* l) {
  __builtin_amdgcn_global_load_async_to_lds_b128((as1_v4i)g, (as3_v4i)l, 0, 0);
}
static __device__ __forceinline__ void async_wait() {
  __builtin_amdgcn_s_wait_asynccnt(0);
}
#else
static __device__ __forceinline__ void async_wait() {}
#endif

static constexpr int Bb = 8, Nn = 4096, Cc = 768, Hh = 12, Dd = 64;
static constexpr int BN = Bb * Nn;           // 32768 rows
static constexpr int C3 = 3 * Cc;            // 2304

// ---------------------------------------------------------------------------
// fp32 -> bf16 convert (prep, memory bound)
// ---------------------------------------------------------------------------
__global__ __launch_bounds__(256) void convert_f32_bf16(const float* __restrict__ in,
                                                        bf16_t* __restrict__ out, size_t n) {
  size_t i = ((size_t)blockIdx.x * 256 + threadIdx.x) * 4;
  if (i + 3 < n) {
    float4 f = *(const float4*)&in[i];
    out[i + 0] = (bf16_t)f.x; out[i + 1] = (bf16_t)f.y;
    out[i + 2] = (bf16_t)f.z; out[i + 3] = (bf16_t)f.w;
  }
}

// ---------------------------------------------------------------------------
// Weight fusion: Weff[seg*C + j, t] = sum_c Wseg[j,c] * Wqkv[seg*C + c, t]
// ---------------------------------------------------------------------------
__global__ __launch_bounds__(256) void fuse_weights(const float* __restrict__ Wqkv,
                                                    const float* __restrict__ Wq,
                                                    const float* __restrict__ Wk,
                                                    const float* __restrict__ Wv,
                                                    bf16_t* __restrict__ Weff) {
  size_t idx = (size_t)blockIdx.x * 256 + threadIdx.x;   // seg*C*C + j*C + t
  int t   = (int)(idx % Cc);
  int j   = (int)((idx / Cc) % Cc);
  int seg = (int)(idx / ((size_t)Cc * Cc));
  const float* Ws = (seg == 0) ? Wq : (seg == 1) ? Wk : Wv;
  const float* Wg = Wqkv + (size_t)seg * Cc * Cc;
  float s0 = 0.f, s1 = 0.f, s2 = 0.f, s3 = 0.f;
  #pragma unroll 4
  for (int c = 0; c < Cc; c += 4) {
    s0 += Ws[(size_t)j * Cc + c + 0] * Wg[(size_t)(c + 0) * Cc + t];
    s1 += Ws[(size_t)j * Cc + c + 1] * Wg[(size_t)(c + 1) * Cc + t];
    s2 += Ws[(size_t)j * Cc + c + 2] * Wg[(size_t)(c + 2) * Cc + t];
    s3 += Ws[(size_t)j * Cc + c + 3] * Wg[(size_t)(c + 3) * Cc + t];
  }
  Weff[idx] = (bf16_t)((s0 + s1) + (s2 + s3));
}

// ---------------------------------------------------------------------------
// Double-buffered bf16 WMMA GEMM: C[M,N] = A[M,K] @ Bw[N,K]^T (+bias +epilogue)
// 128x256 block tile, BK=32. 8 waves (2m x 4n), wave tile 64x64 = 4x4 frags:
// 16 WMMA per 16 ds_load_b128 per k-step, one barrier per k-step.
// Staging via GLOBAL_LOAD_ASYNC_TO_LDS_B128 (ASYNCcnt), prefetch of tile k+1
// overlaps the WMMAs of tile k.
// EPI==0: qkv epilogue (3-segment bias, elu+1 on cols < 2*C), bf16 out
// EPI==1: bias0 + fp32 out
// ---------------------------------------------------------------------------
static constexpr int GBM = 128, GBN = 256, GBK = 32, LDSR = GBK + 8;  // 40-elem rows

template <int EPI>
__global__ __launch_bounds__(256) void gemm_db(const bf16_t* __restrict__ A,
                                               const bf16_t* __restrict__ Bw,
                                               const float* __restrict__ bias0,
                                               const float* __restrict__ bias1,
                                               const float* __restrict__ bias2,
                                               void* __restrict__ Cout,
                                               int M, int N, int K) {
  __shared__ bf16_t As[2][GBM * LDSR];   // 2 x 10240 B
  __shared__ bf16_t Bs[2][GBN * LDSR];   // 2 x 20480 B
  const int tid  = threadIdx.x;
  const int lane = tid & 31, wave = tid >> 5;
  const int wm = wave >> 2, wn = wave & 3;             // 2 x 4 wave grid
  const int m0 = blockIdx.x * GBM;
  const int n0 = blockIdx.y * GBN;
  const int lrow = lane & 15;
  const int ka = (lane >> 4) * 8;    // unused; A and B frags both load at kb
  const int kb = (lane >> 4) * 16;   // A/B-frag k base (contiguous k = 16h + s)
  const int nk = K / GBK;

  v8f acc[4][4] = {};

  auto stage = [&](int bsel, int k0) {
    #pragma unroll
    for (int i = 0; i < 2; ++i) {                      // A: 512 16B chunks
      int c = tid * 2 + i, r = c >> 2, col = (c & 3) * 8;
#if USE_ASYNC_LDS
      async_cp16(&A[(size_t)(m0 + r) * K + k0 + col], &As[bsel][r * LDSR + col]);
#else
      *(v8bf*)&As[bsel][r * LDSR + col] = *(const v8bf*)&A[(size_t)(m0 + r) * K + k0 + col];
#endif
    }
    #pragma unroll
    for (int i = 0; i < 4; ++i) {                      // B: 1024 16B chunks
      int c = tid * 4 + i, r = c >> 2, col = (c & 3) * 8;
#if USE_ASYNC_LDS
      async_cp16(&Bw[(size_t)(n0 + r) * K + k0 + col], &Bs[bsel][r * LDSR + col]);
#else
      *(v8bf*)&Bs[bsel][r * LDSR + col] = *(const v8bf*)&Bw[(size_t)(n0 + r) * K + k0 + col];
#endif
    }
  };
  auto wait_stage = [&]() {
    async_wait();
    __syncthreads();
  };

  stage(0, 0);
  wait_stage();

  for (int kt = 0; kt < nk; ++kt) {
    const int cur = kt & 1;
    if (kt + 1 < nk) stage(1 - cur, (kt + 1) * GBK);   // overlap with WMMA below

    v16bf af[4], bfm[4];
    #pragma unroll
    for (int mf = 0; mf < 4; ++mf) {
      const bf16_t* p = &As[cur][(wm * 64 + mf * 16 + lrow) * LDSR];
      af[mf] = cat8(*(const v8bf*)&p[kb], *(const v8bf*)&p[kb + 8]);
    }
    #pragma unroll
    for (int nf = 0; nf < 4; ++nf) {
      const bf16_t* p = &Bs[cur][(wn * 64 + nf * 16 + lrow) * LDSR];
      bfm[nf] = cat8(*(const v8bf*)&p[kb], *(const v8bf*)&p[kb + 8]);
    }
    #pragma unroll
    for (int mf = 0; mf < 4; ++mf)
      #pragma unroll
      for (int nf = 0; nf < 4; ++nf)
        acc[mf][nf] = WMMA_BF16(af[mf], bfm[nf], acc[mf][nf]);

    if (kt + 1 < nk) wait_stage();
  }

  const int chalf = (lane >> 4) * 8;                   // C/D frag: VGPR r -> rows r, r+8
  #pragma unroll
  for (int mf = 0; mf < 4; ++mf)
    #pragma unroll
    for (int nf = 0; nf < 4; ++nf)
      #pragma unroll
      for (int r = 0; r < 8; ++r) {
        int gm = m0 + wm * 64 + mf * 16 + r + chalf;
        int gn = n0 + wn * 64 + nf * 16 + (lane & 15);
        float v = acc[mf][nf][r];
        if (EPI == 0) {
          float bia = (gn < Cc) ? bias0[gn] : (gn < 2 * Cc) ? bias1[gn - Cc] : bias2[gn - 2 * Cc];
          v += bia;
          if (gn < 2 * Cc) v = (v > 0.f) ? (v + 1.f) : __expf(v);   // elu(x)+1 for Q,K only
          ((bf16_t*)Cout)[(size_t)gm * N + gn] = (bf16_t)v;
        } else {
          v += bias0[gn];
          ((float*)Cout)[(size_t)gm * N + gn] = v;
        }
      }
}

// ---------------------------------------------------------------------------
// Per-(b,h): KV[m,d] = sum_s V[s,m]*K[s,d]  (= V^T @ K, 64x64, K-dim 4096)
// and Ksum[d] = sum_s K[s,d]. Operands transposed into LDS feature-major
// (transpose also feeds the Ksum reduction, so LDS staging stays).
// 8 waves: 2(m) x 4(d); each wave 32x16 = 2 WMMA frags.
// ---------------------------------------------------------------------------
__global__ __launch_bounds__(256) void kv_kernel(const bf16_t* __restrict__ qkv,
                                                 bf16_t* __restrict__ KVout,
                                                 float* __restrict__ Ksum) {
  const int bh = blockIdx.x;                 // b*H + h
  const int b = bh / Hh, h = bh % Hh;
  __shared__ bf16_t Kt[64 * 40];             // [d][s]
  __shared__ bf16_t Vt[64 * 40];             // [m][s]
  const int tid = threadIdx.x, lane = tid & 31, wave = tid >> 5;
  const int mb = (wave >> 2) * 32;
  const int db = (wave & 3) * 16;
  v8f acc[2] = {};
  float ks0 = 0.f, ks1 = 0.f, ks2 = 0.f, ks3 = 0.f;

  const bf16_t* Kbase = qkv + (size_t)b * Nn * C3 + Cc     + h * Dd;
  const bf16_t* Vbase = qkv + (size_t)b * Nn * C3 + 2 * Cc + h * Dd;
  const int lrow = lane & 15;
  const int ka = (lane >> 4) * 8;
  const int kb = (lane >> 4) * 16;

  for (int s0 = 0; s0 < Nn; s0 += 32) {
    {   // stage + transpose: 32 s-rows x 64 features, one 16B chunk per thread
      int s = tid >> 3, d8 = (tid & 7) * 8;
      v8bf kv8 = *(const v8bf*)&Kbase[(size_t)(s0 + s) * C3 + d8];
      v8bf vv8 = *(const v8bf*)&Vbase[(size_t)(s0 + s) * C3 + d8];
      #pragma unroll
      for (int j = 0; j < 8; ++j) {
        Kt[(d8 + j) * 40 + s] = kv8[j];
        Vt[(d8 + j) * 40 + s] = vv8[j];
      }
    }
    __syncthreads();
    v16bf bfr;
    {
      const bf16_t* p = &Kt[(db + lrow) * 40];
      bfr = cat8(*(const v8bf*)&p[kb], *(const v8bf*)&p[kb + 8]);
    }
    #pragma unroll
    for (int mf = 0; mf < 2; ++mf) {
      const bf16_t* p = &Vt[(mb + mf * 16 + lrow) * 40];
      v16bf afr = cat8(*(const v8bf*)&p[kb], *(const v8bf*)&p[kb + 8]);
      acc[mf] = WMMA_BF16(afr, bfr, acc[mf]);
    }
    if (tid < 64) {                         // 4 independent partials: no serial chain
      #pragma unroll
      for (int s = 0; s < 32; s += 4) {
        ks0 += (float)Kt[tid * 40 + s + 0];
        ks1 += (float)Kt[tid * 40 + s + 1];
        ks2 += (float)Kt[tid * 40 + s + 2];
        ks3 += (float)Kt[tid * 40 + s + 3];
      }
    }
    __syncthreads();
  }

  const int chalf = (lane >> 4) * 8;
  bf16_t* kvdst = KVout + (size_t)bh * Dd * Dd;
  #pragma unroll
  for (int mf = 0; mf < 2; ++mf)
    #pragma unroll
    for (int r = 0; r < 8; ++r) {
      int m = mb + mf * 16 + r + chalf;
      int d = db + (lane & 15);
      kvdst[m * Dd + d] = (bf16_t)acc[mf][r];
    }
  if (tid < 64) Ksum[bh * Dd + tid] = (ks0 + ks1) + (ks2 + ks3);
}

// ---------------------------------------------------------------------------
// Per-(b,h, 256-row block): out[l,m] = Z[l] * sum_d Q[l,d]*KV[m,d]
// Z[l] = 1/(Q[l,:]·Ksum + eps). KV's [m][d] layout == WMMA B-frag layout.
// Q block (256x64) async-staged into LDS once (coalesced), reused by the
// Z pass and the A-fragments. 8 waves x 32 rows; 2x4 frags, K=64 in 2 steps.
// ---------------------------------------------------------------------------
__global__ __launch_bounds__(256) void attn_out_kernel(const bf16_t* __restrict__ qkv,
                                                       const bf16_t* __restrict__ KV,
                                                       const float* __restrict__ Ksum,
                                                       bf16_t* __restrict__ out) {
  const int bh = blockIdx.x, lblk = blockIdx.y;
  const int b = bh / Hh, h = bh % Hh;
  __shared__ bf16_t Qs[256 * 72];            // [l][d], 36 KB
  __shared__ bf16_t KVs[64 * 72];            // [m][d], 9 KB
  __shared__ float  ks[64];
  __shared__ float  zs[256];
  const int tid = threadIdx.x, lane = tid & 31, wave = tid >> 5;
  const int l0 = lblk * 256;
  const bf16_t* Qbase = qkv + (size_t)b * Nn * C3 + h * Dd;

  #pragma unroll
  for (int i = 0; i < 8; ++i) {              // stage 256x64 bf16 Q block (2048 chunks)
    int c = tid * 8 + i;
    int r = c >> 3, d8 = (c & 7) * 8;
#if USE_ASYNC_LDS
    async_cp16(&Qbase[(size_t)(l0 + r) * C3 + d8], &Qs[r * 72 + d8]);
#else
    *(v8bf*)&Qs[r * 72 + d8] = *(const v8bf*)&Qbase[(size_t)(l0 + r) * C3 + d8];
#endif
  }
  #pragma unroll
  for (int i = 0; i < 2; ++i) {              // stage 64x64 bf16 KV tile (512 chunks)
    int c = tid * 2 + i;
    int m = c >> 3, d8 = (c & 7) * 8;
#if USE_ASYNC_LDS
    async_cp16(&KV[(size_t)bh * Dd * Dd + m * Dd + d8], &KVs[m * 72 + d8]);
#else
    *(v8bf*)&KVs[m * 72 + d8] = *(const v8bf*)&KV[(size_t)bh * Dd * Dd + m * Dd + d8];
#endif
  }
  if (tid < 64) ks[tid] = Ksum[bh * Dd + tid];
  async_wait();
  __syncthreads();

  {   // normalizer per row (4 independent FMA chains), Q read from LDS
    const bf16_t* q = &Qs[tid * 72];
    float d0 = 0.f, d1 = 0.f, d2 = 0.f, d3 = 0.f;
    #pragma unroll
    for (int d = 0; d < Dd; d += 4) {
      d0 += (float)q[d + 0] * ks[d + 0];
      d1 += (float)q[d + 1] * ks[d + 1];
      d2 += (float)q[d + 2] * ks[d + 2];
      d3 += (float)q[d + 3] * ks[d + 3];
    }
    zs[tid] = 1.f / (((d0 + d1) + (d2 + d3)) + 1e-6f);
  }
  __syncthreads();

  v8f acc[2][4] = {};
  const int lrow = lane & 15;
  const int ka = (lane >> 4) * 8;
  const int kb = (lane >> 4) * 16;
  #pragma unroll
  for (int kc = 0; kc < 2; ++kc) {
    v16bf afr[2], bfr[4];
    #pragma unroll
    for (int mf = 0; mf < 2; ++mf) {
      const bf16_t* q = &Qs[(wave * 32 + mf * 16 + lrow) * 72 + kc * 32];
      afr[mf] = cat8(*(const v8bf*)&q[kb], *(const v8bf*)&q[kb + 8]);
    }
    #pragma unroll
    for (int nf = 0; nf < 4; ++nf) {
      const bf16_t* p = &KVs[(nf * 16 + lrow) * 72 + kc * 32];
      bfr[nf] = cat8(*(const v8bf*)&p[kb], *(const v8bf*)&p[kb + 8]);
    }
    #pragma unroll
    for (int mf = 0; mf < 2; ++mf)
      #pragma unroll
      for (int nf = 0; nf < 4; ++nf)
        acc[mf][nf] = WMMA_BF16(afr[mf], bfr[nf], acc[mf][nf]);
  }

  const int chalf = (lane >> 4) * 8;
  #pragma unroll
  for (int mf = 0; mf < 2; ++mf)
    #pragma unroll
    for (int nf = 0; nf < 4; ++nf)
      #pragma unroll
      for (int r = 0; r < 8; ++r) {
        int lloc = wave * 32 + mf * 16 + r + chalf;
        int m    = nf * 16 + (lane & 15);
        float v  = acc[mf][nf][r] * zs[lloc];
        out[((size_t)(b * Nn + l0 + lloc)) * Cc + h * Dd + m] = (bf16_t)v;
      }
}

// ---------------------------------------------------------------------------
extern "C" void kernel_launch(void* const* d_in, const int* in_sizes, int n_in,
                              void* d_out, int out_size, void* d_ws, size_t ws_size,
                              hipStream_t stream) {
  (void)in_sizes; (void)n_in; (void)out_size; (void)ws_size;
  const float* x    = (const float*)d_in[0];
  const float* Wqkv = (const float*)d_in[1];
  const float* Wq   = (const float*)d_in[2];
  const float* bq   = (const float*)d_in[3];
  const float* Wk   = (const float*)d_in[4];
  const float* bk   = (const float*)d_in[5];
  const float* Wv   = (const float*)d_in[6];
  const float* bv   = (const float*)d_in[7];
  const float* Wo   = (const float*)d_in[8];
  const float* bo   = (const float*)d_in[9];
  float* outp = (float*)d_out;

  char* ws = (char*)d_ws;
  bf16_t* x_bf   = (bf16_t*)ws;  ws += (size_t)BN * Cc * 2;     // 50.3 MB
  bf16_t* Weff   = (bf16_t*)ws;  ws += (size_t)C3 * Cc * 2;     //  3.5 MB
  bf16_t* Wo_bf  = (bf16_t*)ws;  ws += (size_t)Cc * Cc * 2;     //  1.2 MB
  bf16_t* qkvbuf = (bf16_t*)ws;  ws += (size_t)BN * C3 * 2;     //  151 MB
  bf16_t* KVbuf  = (bf16_t*)ws;  ws += (size_t)Bb * Hh * Dd * Dd * 2;
  float*  Ksum   = (float*)ws;   ws += (size_t)Bb * Hh * Dd * 4;
  bf16_t* attn   = (bf16_t*)ws;  ws += (size_t)BN * Cc * 2;     // 50.3 MB

  size_t nx = (size_t)BN * Cc;
  convert_f32_bf16<<<(unsigned)((nx / 4 + 255) / 256), 256, 0, stream>>>(x, x_bf, nx);
  size_t nw = (size_t)Cc * Cc;
  convert_f32_bf16<<<(unsigned)((nw / 4 + 255) / 256), 256, 0, stream>>>(Wo, Wo_bf, nw);

  fuse_weights<<<(unsigned)((size_t)3 * Cc * Cc / 256), 256, 0, stream>>>(Wqkv, Wq, Wk, Wv, Weff);

  gemm_db<0><<<dim3(BN / GBM, C3 / GBN), 256, 0, stream>>>(
      x_bf, Weff, bq, bk, bv, qkvbuf, BN, C3, Cc);

  kv_kernel<<<Bb * Hh, 256, 0, stream>>>(qkvbuf, KVbuf, Ksum);

  attn_out_kernel<<<dim3(Bb * Hh, Nn / 256), 256, 0, stream>>>(qkvbuf, KVbuf, Ksum, attn);

  gemm_db<1><<<dim3(BN / GBM, Cc / GBN), 256, 0, stream>>>(
      attn, Wo_bf, bo, nullptr, nullptr, outp, BN, Cc, Cc);
}
